// EdgeMLP_61598420959300
// MI455X (gfx1250) — matrix operand order repaired and checked by reading, compiled, not run on hardware
//
#include <hip/hip_runtime.h>

typedef __attribute__((ext_vector_type(16))) _Float16 v16h;
typedef __attribute__((ext_vector_type(8)))  float    v8f;

#define HID 10
#define IN_DIM 13

// D = A x B + C with A = W1^T (16x32, per-wave constant), B = feat^T (32x16,
// column = edge, K = feature). Lanes 0-15 carry K=0..15 (tile 0 edges), lanes
// 16-31 carry K=16..31 (tile 1 edges). A_lo selects K=0..12, A_hi selects
// K=16..28, so one B operand feeds two 16-edge tiles -> 32 edges/iteration.
__global__ __launch_bounds__(256) void edge_mlp_wmma_kernel(
    const float* __restrict__ x,        // [N,6]
    const int*   __restrict__ eidx,     // int64 viewed as int pairs: [2,E]
    const float* __restrict__ ea,       // [E]
    const float* __restrict__ eaT,      // [E]
    const float* __restrict__ W1,       // [13,10]
    const float* __restrict__ b1,       // [10]
    const float* __restrict__ W2,       // [10,2]
    const float* __restrict__ b2,       // [2]
    float* __restrict__ out,            // [E]
    int nEdges)
{
    const int lane = threadIdx.x & 31;
    const int half = lane >> 4;
    const int col  = lane & 15;         // A: hidden-unit row j ; D: edge column

    // ---- per-wave constant fragments ----
    // A-layout (16-bit 16x32, contiguous K): lanes 0-15 row M=lane, halfs
    // 0-15 = K0-15; lanes 16-31 row M=lane-16, halfs 0-15 = K16-31.
    float wv[16];
#pragma unroll
    for (int i = 0; i < 16; ++i) {
        const int k = i;                                  // feature index 0..15
        wv[i] = (col < HID && k < IN_DIM) ? W1[k * HID + col] : 0.f;
    }
    v16h a_lo = {}, a_hi = {};
#pragma unroll
    for (int i = 0; i < 16; ++i) {
        a_lo[i] = half ? (_Float16)0.f : (_Float16)wv[i]; // W1^T at K=0..12
        a_hi[i] = half ? (_Float16)wv[i] : (_Float16)0.f; // W1^T at K=16..28
    }

    // D-layout: VGPR k -> row j = k + 8*half, col = lane&15.
    // 0.5 softmax-average factor folded into the layer-2 weight diff.
    v8f cbias;
    float w2v[8];
#pragma unroll
    for (int k = 0; k < 8; ++k) {
        const int j = k + 8 * half;
        cbias[k] = (j < HID) ? b1[j] : 0.f;
        w2v[k]   = (j < HID) ? 0.5f * (W2[2 * j] - W2[2 * j + 1]) : 0.f;
    }
    const float b2d = b2[0] - b2[1];
    const float kInf = __builtin_inff();

    const int nGroups = (nEdges + 31) >> 5;               // 32 edges per group
    const int wavesPerBlock = blockDim.x >> 5;
    const int wave   = blockIdx.x * wavesPerBlock + (threadIdx.x >> 5);
    const int nWaves = gridDim.x * wavesPerBlock;

    for (int g = wave; g < nGroups; g += nWaves) {
        const int base = g << 5;
        int m = base + lane;
        if (m >= nEdges) m = nEdges - 1;

        // Prefetch this wave's next group of streamed data (global_prefetch_b8)
        {
            const int ng = g + nWaves;
            if (ng < nGroups) {
                int nm = (ng << 5) + lane;
                if (nm < nEdges) {
                    __builtin_prefetch(&eidx[2 * (size_t)nm], 0, 1);
                    __builtin_prefetch(&eidx[2 * ((size_t)nEdges + (size_t)nm)], 0, 1);
                    __builtin_prefetch(&ea[nm], 0, 1);
                    __builtin_prefetch(&eaT[nm], 0, 1);
                }
            }
        }

        const int src = eidx[2 * (size_t)m];                        // low dword of int64
        const int tgt = eidx[2 * ((size_t)nEdges + (size_t)m)];

        const float2* ps = (const float2*)(x + 6 * (size_t)src);    // 8B aligned rows
        const float2* pt = (const float2*)(x + 6 * (size_t)tgt);
        const float2 s01 = ps[0], s23 = ps[1], s45 = ps[2];
        const float2 t01 = pt[0], t23 = pt[1], t45 = pt[2];
        const float eav = ea[m];
        const float etv = eaT[m];

        // B fragments: this lane's edge features in halfs 0..12 (uniform code,
        // no lane-dependent selects). fwd = [xs, xt, ea], rev = [xt, xs, eaT].
        v16h bf = {}, br = {};
        bf[0]  = (_Float16)s01.x;  br[0]  = (_Float16)t01.x;
        bf[1]  = (_Float16)s01.y;  br[1]  = (_Float16)t01.y;
        bf[2]  = (_Float16)s23.x;  br[2]  = (_Float16)t23.x;
        bf[3]  = (_Float16)s23.y;  br[3]  = (_Float16)t23.y;
        bf[4]  = (_Float16)s45.x;  br[4]  = (_Float16)t45.x;
        bf[5]  = (_Float16)s45.y;  br[5]  = (_Float16)t45.y;
        bf[6]  = (_Float16)t01.x;  br[6]  = (_Float16)s01.x;
        bf[7]  = (_Float16)t01.y;  br[7]  = (_Float16)s01.y;
        bf[8]  = (_Float16)t23.x;  br[8]  = (_Float16)s23.x;
        bf[9]  = (_Float16)t23.y;  br[9]  = (_Float16)s23.y;
        bf[10] = (_Float16)t45.x;  br[10] = (_Float16)s45.x;
        bf[11] = (_Float16)t45.y;  br[11] = (_Float16)s45.y;
        bf[12] = (_Float16)eav;    br[12] = (_Float16)etv;
        // halfs 13..15 stay zero (K padding)

        // H^T = W1^T @ feat^T + b1 : tile0 (edges base..base+15) via A_lo,
        // tile1 (edges base+16..base+31) via A_hi, same B operand.
        v8f hf0 = __builtin_amdgcn_wmma_f32_16x16x32_f16(
            false, a_lo, false, bf, (short)0, cbias, false, false);
        v8f hr0 = __builtin_amdgcn_wmma_f32_16x16x32_f16(
            false, a_lo, false, br, (short)0, cbias, false, false);
        v8f hf1 = __builtin_amdgcn_wmma_f32_16x16x32_f16(
            false, a_hi, false, bf, (short)0, cbias, false, false);
        v8f hr1 = __builtin_amdgcn_wmma_f32_16x16x32_f16(
            false, a_hi, false, br, (short)0, cbias, false, false);

        // Layer 2 + softmax collapse: logit-diff(edge) =
        //   sum_j (relu(hf_j)+relu(hr_j)) * 0.5*(W2[j,0]-W2[j,1]) + (b2[0]-b2[1])
        // relu as single v_med3_f32 clamp to [0, +inf). j = k + 8*half lives in
        // this lane's VGPRs -> in-register FMA chain, then ONE xor-16 shuffle.
        float p0 = 0.f, p1 = 0.f;
#pragma unroll
        for (int k = 0; k < 8; ++k) {
            const float rf0 = __builtin_amdgcn_fmed3f(hf0[k], 0.f, kInf);
            const float rr0 = __builtin_amdgcn_fmed3f(hr0[k], 0.f, kInf);
            const float rf1 = __builtin_amdgcn_fmed3f(hf1[k], 0.f, kInf);
            const float rr1 = __builtin_amdgcn_fmed3f(hr1[k], 0.f, kInf);
            p0 += (rf0 + rr0) * w2v[k];
            p1 += (rf1 + rr1) * w2v[k];
        }
        p0 += __shfl_xor(p0, 16);
        p1 += __shfl_xor(p1, 16);

        const float s = (half ? p1 : p0) + b2d;           // o0 - o1 (per own edge)
        // sigmoid(-s): v_exp_f32 + v_rcp_f32, no refinement chain
        const float e = __builtin_amdgcn_exp2f(s * 1.44269504088896340736f);
        const float y = __builtin_amdgcn_rcpf(1.f + e);   // softmax[:,1]

        const int eo = base + lane;
        if (eo < nEdges) out[eo] = y;                     // coalesced 32-lane store
    }
}

extern "C" void kernel_launch(void* const* d_in, const int* in_sizes, int n_in,
                              void* d_out, int out_size, void* d_ws, size_t ws_size,
                              hipStream_t stream) {
    const float* x   = (const float*)d_in[0];
    const int*   ei  = (const int*)d_in[1];   // int64 buffer, low-dword reads
    const float* ea  = (const float*)d_in[2];
    const float* eaT = (const float*)d_in[3];
    const float* W1  = (const float*)d_in[4];
    const float* b1  = (const float*)d_in[5];
    const float* W2  = (const float*)d_in[6];
    const float* b2  = (const float*)d_in[7];
    float* out = (float*)d_out;

    const int nEdges = in_sizes[2];           // edge_attr is [E]
    const int groups = (nEdges + 31) >> 5;
    const int wavesPerBlock = 256 / 32;
    int blocks = (groups + wavesPerBlock - 1) / wavesPerBlock;
    if (blocks > 4096) blocks = 4096;         // persistent grid-stride waves
    if (blocks < 1) blocks = 1;

    edge_mlp_wmma_kernel<<<blocks, 256, 0, stream>>>(
        x, ei, ea, eaT, W1, b1, W2, b2, out, nEdges);
}
